// CrossAttention_42941083025507
// MI455X (gfx1250) — matrix operand order repaired and checked by reading, not compiled, not run on hardware
//
#include <hip/hip_runtime.h>
#include <hip/hip_bf16.h>

// ---------------------------------------------------------------------------
// CDNA5 (gfx1250) fused cross-attention:
//   phase 1: QKV 1x1-conv projections via v_wmma_f32_16x16x32_bf16
//   phase 2: flash-attention (online softmax) via the same WMMA op
// wave32; K/V tiles staged with GLOBAL_LOAD_ASYNC_TO_LDS_B128 when available.
// ---------------------------------------------------------------------------

#define NTOK 4096   // H*W tokens per batch
#define CDIM 128    // channels
#define NB   4      // batch

#define BQ 64       // query rows per workgroup (4 waves x 16 rows)
#define BK 64       // key tile staged in LDS
#define XP 136      // LDS pitch (bf16 elems) for [64][128] tiles (16B-aligned rows)
#define WP 136      // LDS pitch for weight tile
#define KP 136      // LDS pitch for K tile rows (len 128)
#define VP 72       // LDS pitch for V tile rows (len 64)
#define PP 72       // LDS pitch for P tile rows (len 64)

typedef __attribute__((ext_vector_type(16))) __bf16 v16bf;
typedef __attribute__((ext_vector_type(8)))  float  v8f;
typedef __attribute__((ext_vector_type(4)))  int    v4i;

struct __attribute__((aligned(16))) U4 { unsigned int x, y, z, w; };

union Frag {            // 32 bytes per lane: one 16x32 bf16 A/B fragment slice
  v16bf v;
  unsigned short u[16];
  U4 q[2];
};

__device__ __forceinline__ unsigned short f2bf(float f) {
  unsigned int u = __float_as_uint(f);
  unsigned int r = u + 0x7FFFu + ((u >> 16) & 1u);   // round-to-nearest-even
  return (unsigned short)(r >> 16);
}

// packed 2x f32 -> bf16 (hardware v_cvt_pk_bf16_f32 when available)
__device__ __forceinline__ unsigned int f2bf2(float a, float b) {
#if __has_builtin(__builtin_amdgcn_cvt_pk_bf16_f32)
  typedef __attribute__((ext_vector_type(2))) __bf16 v2bf;
  v2bf t = __builtin_amdgcn_cvt_pk_bf16_f32(a, b);
  unsigned int r;
  __builtin_memcpy(&r, &t, 4);
  return r;
#else
  return (unsigned int)f2bf(a) | ((unsigned int)f2bf(b) << 16);
#endif
}

#if __has_builtin(__builtin_amdgcn_global_load_async_to_lds_b128) && \
    __has_builtin(__builtin_amdgcn_s_wait_asynccnt)
#define HAVE_ASYNC_LDS 1
typedef __attribute__((address_space(1))) v4i* g_v4i_ptr;
typedef __attribute__((address_space(3))) v4i* l_v4i_ptr;
__device__ __forceinline__ void async_copy16(const void* g, void* l) {
  __builtin_amdgcn_global_load_async_to_lds_b128(
      (g_v4i_ptr)g, (l_v4i_ptr)l, 0, 0);
}
#endif

// ---------------------------------------------------------------------------
// Phase 1: Q/K/V projection.  blockIdx.z selects which projection.
//   Q,K -> ws as bf16 [B][N][C]   (row = token, contiguous channels)
//   V   -> ws as bf16 [B][C][N]   (transposed: contiguous tokens per channel)
// ---------------------------------------------------------------------------
__global__ __launch_bounds__(128) void qkv_proj_kernel(
    const float* __restrict__ x1, const float* __restrict__ x2,
    const float* __restrict__ wq, const float* __restrict__ bq,
    const float* __restrict__ wk, const float* __restrict__ bk,
    const float* __restrict__ wv, const float* __restrict__ bv,
    unsigned short* __restrict__ Qg, unsigned short* __restrict__ Kg,
    unsigned short* __restrict__ Vt)
{
  __shared__ unsigned short Xs[64 * XP];    // token-major x tile (transposed in)
  __shared__ unsigned short Ws[CDIM * WP];  // weight [o][c]

  const int n0    = blockIdx.x * 64;
  const int b     = blockIdx.y;
  const int which = blockIdx.z;             // 0=Q 1=K 2=V
  const float* xin  = (which == 0) ? x1 : x2;
  const float* w    = (which == 0) ? wq : (which == 1) ? wk : wv;
  const float* bias = (which == 0) ? bq : (which == 1) ? bk : bv;

  const int tid = threadIdx.x;

  // x[b][c][n0+n] -> Xs[n][c]: convert channel PAIRS, one u32 LDS store each.
#pragma unroll 4
  for (int it = 0; it < 32; ++it) {
    int p = it * 128 + tid;                 // 4096 pairs
    int n = p & 63;
    int c = (p >> 6) * 2;
    const float* s0 = xin + ((size_t)b * CDIM + c) * NTOK + n0 + n;
    *(unsigned int*)&Xs[n * XP + c] = f2bf2(s0[0], s0[NTOK]);
  }
  // w[o][c] -> Ws[o][c]: contiguous pairs via float2 loads.
  {
    const float2* wf2 = (const float2*)w;
#pragma unroll 4
    for (int it = 0; it < 64; ++it) {
      int p = it * 128 + tid;               // 8192 pairs
      int o = p >> 6;
      int c = (p & 63) * 2;
      float2 t = wf2[p];
      *(unsigned int*)&Ws[o * WP + c] = f2bf2(t.x, t.y);
    }
  }
  __syncthreads();

  const int wave  = tid >> 5;
  const int lane  = tid & 31;
  const int lhalf = lane >> 4;
  const int l16   = lane & 15;
  const int h8    = lhalf * 8;
  const int h16   = lhalf * 16;
  const int rowbase = wave * 16;

  // A fragments: 16 token rows x 128 channels = 4 K-steps of 32
  Frag a[4];
  {
    const unsigned short* src = &Xs[(rowbase + l16) * XP];
#pragma unroll
    for (int cs = 0; cs < 4; ++cs) {
      a[cs].q[0] = *(const U4*)(src + cs * 32 + h8);
      a[cs].q[1] = *(const U4*)(src + cs * 32 + 16 + h8);
    }
  }

#pragma unroll
  for (int ot = 0; ot < 8; ++ot) {
    v8f acc = {};
#pragma unroll
    for (int cs = 0; cs < 4; ++cs) {
      Frag bf_;
      const unsigned short* src = &Ws[(ot * 16 + l16) * WP + cs * 32 + h8];
      bf_.q[0] = *(const U4*)(src);
      bf_.q[1] = *(const U4*)(src + 16);
      acc = __builtin_amdgcn_wmma_f32_16x16x32_bf16(
          false, a[cs].v, false, bf_.v, (short)0, acc, false, false);
    }
    const int   o   = ot * 16 + l16;
    const float bv_ = bias[o];
    if (which == 2) {
      // Vt[b][o][n]: 8 consecutive tokens per lane -> one 16B store
      U4 tmp;
      tmp.x = f2bf2(acc[0] + bv_, acc[1] + bv_);
      tmp.y = f2bf2(acc[2] + bv_, acc[3] + bv_);
      tmp.z = f2bf2(acc[4] + bv_, acc[5] + bv_);
      tmp.w = f2bf2(acc[6] + bv_, acc[7] + bv_);
      size_t off = ((size_t)b * CDIM + o) * NTOK + n0 + rowbase + h8;
      *(U4*)(Vt + off) = tmp;
    } else {
      unsigned short* dst = (which == 0) ? Qg : Kg;
#pragma unroll
      for (int r = 0; r < 8; ++r) {
        int n = n0 + rowbase + h8 + r;
        dst[((size_t)b * NTOK + n) * CDIM + o] = f2bf(acc[r] + bv_);
      }
    }
  }
}

// ---------------------------------------------------------------------------
// Phase 2: flash attention + residual.
// 4 waves per WG; wave w owns query rows [q0 + 16w, q0 + 16w + 16).
// ---------------------------------------------------------------------------
__global__ __launch_bounds__(128) void attn_kernel(
    const unsigned short* __restrict__ Qg,
    const unsigned short* __restrict__ Kg,
    const unsigned short* __restrict__ Vt,
    const float* __restrict__ x1, float* __restrict__ out)
{
  __shared__ unsigned short Ks[BK * KP];       // [key][c]
  __shared__ unsigned short Vs[CDIM * VP];     // [c][key]  (transposed tile)
  __shared__ unsigned short Ps[4 * 16 * PP];   // per-wave softmaxed P tile

  const int q0   = blockIdx.x * BQ;
  const int b    = blockIdx.y;
  const int tid  = threadIdx.x;
  const int wave = tid >> 5;
  const int lane = tid & 31;
  const int lhalf = lane >> 4;
  const int l16   = lane & 15;
  const int h8    = lhalf * 8;
  const int h16   = lhalf * 16;

  // Q fragments for this wave's 16 rows (kept in registers for whole kernel)
  Frag qf[4];
  {
    const unsigned short* src =
        Qg + ((size_t)b * NTOK + q0 + wave * 16 + l16) * CDIM;
#pragma unroll
    for (int cs = 0; cs < 4; ++cs) {
      qf[cs].q[0] = *(const U4*)(src + cs * 32 + h8);
      qf[cs].q[1] = *(const U4*)(src + cs * 32 + 16 + h8);
    }
  }

  float m_r[8], l_r[8];
  v8f o_acc[8];
  {
    v8f z = {};
#pragma unroll
    for (int ct = 0; ct < 8; ++ct) o_acc[ct] = z;
#pragma unroll
    for (int r = 0; r < 8; ++r) { m_r[r] = -1e30f; l_r[r] = 0.0f; }
  }

  const float c1 = 1.4426950408889634f * 0.08838834764831845f; // log2e/sqrt(C)
  const unsigned short* Kbase = Kg + (size_t)b * NTOK * CDIM;
  const unsigned short* Vbase = Vt + (size_t)b * CDIM * NTOK;
  unsigned short* Pw = &Ps[wave * 16 * PP];

  for (int k0 = 0; k0 < NTOK; k0 += BK) {
    // ---- stage K tile [BK][C] and V tile [C][BK] into LDS ----
    // 16B chunk per lane per issue; async straight to LDS on CDNA5.
#pragma unroll
    for (int it = 0; it < 8; ++it) {
      int idx = (it * 128 + tid) * 8;        // bf16 element index, 16B chunks
      int row = idx >> 7;
      int col = idx & 127;
      const void* g = Kbase + (size_t)(k0 + row) * CDIM + col;
      void*       l = &Ks[row * KP + col];
#ifdef HAVE_ASYNC_LDS
      async_copy16(g, l);
#else
      *(U4*)l = *(const U4*)g;
#endif
    }
#pragma unroll
    for (int it = 0; it < 8; ++it) {
      int idx = (it * 128 + tid) * 8;
      int c  = idx >> 6;
      int kk = idx & 63;
      const void* g = Vbase + (size_t)c * NTOK + k0 + kk;
      void*       l = &Vs[c * VP + kk];
#ifdef HAVE_ASYNC_LDS
      async_copy16(g, l);
#else
      *(U4*)l = *(const U4*)g;
#endif
    }
    if (k0 + BK < NTOK) {                    // hint next tiles into cache
      __builtin_prefetch(Kbase + (size_t)(k0 + BK) * CDIM + tid * 64, 0, 1);
      __builtin_prefetch(Vbase + (size_t)tid * NTOK + k0 + BK, 0, 1);
    }
#ifdef HAVE_ASYNC_LDS
    __builtin_amdgcn_s_wait_asynccnt(0);
#endif
    __syncthreads();

    // ---- S = Q * K^T  (4 accumulator tiles covering 64 keys) ----
    v8f s[4];
#pragma unroll
    for (int kt = 0; kt < 4; ++kt) {
      v8f acc = {};
#pragma unroll
      for (int cs = 0; cs < 4; ++cs) {
        Frag bf_;
        const unsigned short* src = &Ks[(kt * 16 + l16) * KP + cs * 32 + h8];
        bf_.q[0] = *(const U4*)(src);
        bf_.q[1] = *(const U4*)(src + 16);
        acc = __builtin_amdgcn_wmma_f32_16x16x32_bf16(
            false, qf[cs].v, false, bf_.v, (short)0, acc, false, false);
      }
      s[kt] = acc;
    }

    // ---- online softmax (per wave-owned row; 16-lane shuffle reductions) ----
#pragma unroll
    for (int r = 0; r < 8; ++r) {
      float mx = fmaxf(fmaxf(s[0][r], s[1][r]), fmaxf(s[2][r], s[3][r]));
#pragma unroll
      for (int off = 8; off >= 1; off >>= 1)
        mx = fmaxf(mx, __shfl_xor(mx, off, 32));
      float mnew = fmaxf(m_r[r], mx);
      float corr = __builtin_exp2f((m_r[r] - mnew) * c1);
      m_r[r] = mnew;

      float p0 = __builtin_exp2f((s[0][r] - mnew) * c1);
      float p1 = __builtin_exp2f((s[1][r] - mnew) * c1);
      float p2 = __builtin_exp2f((s[2][r] - mnew) * c1);
      float p3 = __builtin_exp2f((s[3][r] - mnew) * c1);
      unsigned short* prow = &Pw[(r + lhalf * 8) * PP + l16];
      prow[0]  = f2bf(p0);
      prow[16] = f2bf(p1);
      prow[32] = f2bf(p2);
      prow[48] = f2bf(p3);

      float rs = (p0 + p1) + (p2 + p3);
#pragma unroll
      for (int off = 8; off >= 1; off >>= 1)
        rs += __shfl_xor(rs, off, 32);
      l_r[r] = l_r[r] * corr + rs;

#pragma unroll
      for (int ct = 0; ct < 8; ++ct) o_acc[ct][r] *= corr;
    }

    // ---- O += P * V  (P from per-wave LDS; wave-internal DS ordering) ----
    Frag pf[2];
#pragma unroll
    for (int ks = 0; ks < 2; ++ks) {
      const unsigned short* src = &Pw[l16 * PP + ks * 32 + h8];
      pf[ks].q[0] = *(const U4*)(src);
      pf[ks].q[1] = *(const U4*)(src + 16);
    }
#pragma unroll
    for (int ct = 0; ct < 8; ++ct) {
      v8f acc = o_acc[ct];
#pragma unroll
      for (int ks = 0; ks < 2; ++ks) {
        Frag bf_;
        const unsigned short* src = &Vs[(ct * 16 + l16) * VP + ks * 32 + h8];
        bf_.q[0] = *(const U4*)(src);
        bf_.q[1] = *(const U4*)(src + 16);
        acc = __builtin_amdgcn_wmma_f32_16x16x32_bf16(
            false, pf[ks].v, false, bf_.v, (short)0, acc, false, false);
      }
      o_acc[ct] = acc;
    }
    __syncthreads();
  }

  // ---- epilogue: normalize, add residual x1, store out[b][c][q] ----
  float inv[8];
#pragma unroll
  for (int r = 0; r < 8; ++r) inv[r] = 1.0f / l_r[r];

  const float* x1b  = x1  + (size_t)b * CDIM * NTOK;
  float*       outb = out + (size_t)b * CDIM * NTOK;
  const int qbase = q0 + wave * 16 + h8;     // 8 consecutive q-rows per lane

#pragma unroll
  for (int ct = 0; ct < 8; ++ct) {
    int c = ct * 16 + l16;
    const float* res = x1b  + (size_t)c * NTOK + qbase;
    float*       dst = outb + (size_t)c * NTOK + qbase;
    float4 a0 = *(const float4*)(res);
    float4 a1 = *(const float4*)(res + 4);
    float4 o0, o1;
    o0.x = a0.x + o_acc[ct][0] * inv[0];
    o0.y = a0.y + o_acc[ct][1] * inv[1];
    o0.z = a0.z + o_acc[ct][2] * inv[2];
    o0.w = a0.w + o_acc[ct][3] * inv[3];
    o1.x = a1.x + o_acc[ct][4] * inv[4];
    o1.y = a1.y + o_acc[ct][5] * inv[5];
    o1.z = a1.z + o_acc[ct][6] * inv[6];
    o1.w = a1.w + o_acc[ct][7] * inv[7];
    *(float4*)(dst)     = o0;
    *(float4*)(dst + 4) = o1;
  }
}

// ---------------------------------------------------------------------------
extern "C" void kernel_launch(void* const* d_in, const int* in_sizes, int n_in,
                              void* d_out, int out_size, void* d_ws, size_t ws_size,
                              hipStream_t stream) {
  const float* x1 = (const float*)d_in[0];
  const float* x2 = (const float*)d_in[1];
  const float* wq = (const float*)d_in[2];
  const float* bq = (const float*)d_in[3];
  const float* wk = (const float*)d_in[4];
  const float* bk = (const float*)d_in[5];
  const float* wv = (const float*)d_in[6];
  const float* bv = (const float*)d_in[7];
  float* out = (float*)d_out;

  const size_t per = (size_t)NB * NTOK * CDIM;      // elements per tensor
  unsigned short* Qg = (unsigned short*)d_ws;       // [B][N][C] bf16
  unsigned short* Kg = Qg + per;                    // [B][N][C] bf16
  unsigned short* Vt = Kg + per;                    // [B][C][N] bf16

  dim3 gProj(NTOK / 64, NB, 3);
  qkv_proj_kernel<<<gProj, 128, 0, stream>>>(x1, x2, wq, bq, wk, bk, wv, bv,
                                             Qg, Kg, Vt);

  dim3 gAttn(NTOK / BQ, NB);
  attn_kernel<<<gAttn, 128, 0, stream>>>(Qg, Kg, Vt, x1, out);
}
